// GlobalPointConv_91293824844261
// MI455X (gfx1250) — compile-verified, matrix operand order checked by reading, not run on hardware
//
#include <hip/hip_runtime.h>
#include <hip/hip_fp16.h>
#include <hip/hip_bf16.h>

#define B_    8
#define K_    1024
#define P_    4096
#define N_    16
#define CIN   64
#define CMID  64
#define COUT  128
#define H1_   32
#define H2_   32
#define FH_   256
#define EDIM  (CIN*CMID)   /* 4096 */
#define ROWS  (B_*K_)      /* 8192 */

typedef __attribute__((ext_vector_type(16))) _Float16 v16h;
typedef __attribute__((ext_vector_type(16))) __bf16   v16bf;
typedef __attribute__((ext_vector_type(8)))  float    v8f;
typedef __attribute__((ext_vector_type(4)))  unsigned v4u;
typedef __attribute__((ext_vector_type(8)))  int      v8i;
typedef __attribute__((ext_vector_type(4)))  int      v4i;

#if __has_builtin(__builtin_amdgcn_tensor_load_to_lds)
#define HAVE_TDM 1
#else
#define HAVE_TDM 0
#endif

// Build a 16-bit A/B fragment from LDS that stores each matrix row as 16
// packed K-pairs (u32 = two consecutive K elements). Per the CDNA5 16-bit
// 16x32 layout: lanes 0-15 hold K-pairs {0..3, 8..11}, lanes 16-31 hold
// {4..7, 12..15} (half = lane>>4).
__device__ __forceinline__ v16h ldsFragH(const unsigned* rowBase, int half) {
  union { unsigned u[8]; v16h v; } x;
#pragma unroll
  for (int i = 0; i < 4; ++i) {
    x.u[i]     = rowBase[half*4 + i];
    x.u[4 + i] = rowBase[8 + half*4 + i];
  }
  return x.v;
}
__device__ __forceinline__ v16bf ldsFragBF(const unsigned* rowBase, int half) {
  union { unsigned u[8]; v16bf v; } x;
#pragma unroll
  for (int i = 0; i < 4; ++i) {
    x.u[i]     = rowBase[half*4 + i];
    x.u[4 + i] = rowBase[8 + half*4 + i];
  }
  return x.v;
}

#if HAVE_TDM
// Issue one 2D-tile TDM load (bf16 elements, data_size code 1 = 2 bytes).
// D# bit layout per CDNA5 ISA:
//  g0: [1:0]=count=1, [63:32]=lds_addr, [120:64]=global_addr, [127:126]=type=2
//  g1: [17:16]=data_size, [79:48]=tensor_dim0, [111:80]=tensor_dim1,
//      [127:112]=tile_dim0, [143:128]=tile_dim1, [207:160]=tensor_dim0_stride
__device__ __forceinline__ void tdm_load_2d(unsigned lds_off, const void* gptr,
                                            unsigned tensor_d0, unsigned tensor_d1,
                                            unsigned tile_d0,   unsigned tile_d1,
                                            unsigned stride0) {
  unsigned long long ga = (unsigned long long)(size_t)gptr;
  v4u g0;
  g0[0] = 1u;                                            // count=1
  g0[1] = lds_off;                                       // LDS byte offset
  g0[2] = (unsigned)(ga & 0xFFFFFFFFu);                  // global_addr[31:0]
  g0[3] = (unsigned)((ga >> 32) & 0x01FFFFFFu) | (2u << 30);  // addr[56:32], type=2
  v8i g1;
  g1[0] = (int)(1u << 16);                               // data_size = 2 bytes
  g1[1] = (int)((tensor_d0 & 0xFFFFu) << 16);            // tensor_dim0 lo16
  g1[2] = (int)((tensor_d0 >> 16) | ((tensor_d1 & 0xFFFFu) << 16));
  g1[3] = (int)((tensor_d1 >> 16) | (tile_d0 << 16));    // tile_dim0
  g1[4] = (int)tile_d1;                                  // tile_dim1 (tile_dim2=0)
  g1[5] = (int)stride0;                                  // tensor_dim0_stride lo32
  g1[6] = 0;
  g1[7] = 0;
  v4i g2 = {0, 0, 0, 0};
  v4i g3 = {0, 0, 0, 0};
#if defined(__clang_major__) && __clang_major__ >= 23
  v8i g4 = {0, 0, 0, 0, 0, 0, 0, 0};
  __builtin_amdgcn_tensor_load_to_lds(g0, g1, g2, g3, g4, 0);
#else
  __builtin_amdgcn_tensor_load_to_lds(g0, g1, g2, g3, 0);
#endif
}
#endif

// ---------------------------------------------------------------------------
// Weight prep: fp32 -> bf16, transposed to N-major so K runs contiguously.
// ---------------------------------------------------------------------------
__global__ void prep_weights(const float* __restrict__ wf1,
                             const float* __restrict__ wf2,
                             __hip_bfloat16* __restrict__ Wt1,
                             __hip_bfloat16* __restrict__ Wt2) {
  int i = blockIdx.x * 256 + threadIdx.x;
  const int T1 = EDIM * FH_;
  if (i < T1) {
    int k = i >> 8, n = i & 255;                     // wf1[k][n], k<4096, n<256
    Wt1[(size_t)n * EDIM + k] = __float2bfloat16(wf1[i]);
  } else if (i < T1 + FH_ * COUT) {
    int j = i - T1;
    int k = j >> 7, n = j & 127;                     // wf2[k][n], k<256, n<128
    Wt2[n * FH_ + k] = __float2bfloat16(wf2[j]);
  }
}

// ---------------------------------------------------------------------------
// Top-16 nearest neighbors: one wave32 per (b,k).
// ---------------------------------------------------------------------------
__global__ void topk_kernel(const float* __restrict__ keys,
                            const float* __restrict__ points,
                            const float* __restrict__ mask,
                            int* __restrict__ idxOut) {
  __shared__ float cd[512];
  __shared__ int   ci[512];
  int bk = blockIdx.x;
  int b  = bk >> 10;
  int l  = threadIdx.x;                              // 0..31
  float kx = keys[(size_t)bk*3 + 0];
  float ky = keys[(size_t)bk*3 + 1];
  float kz = keys[(size_t)bk*3 + 2];
  int base = l * 16;
#pragma unroll
  for (int j = 0; j < 16; ++j) { cd[base + j] = 3.0e38f; ci[base + j] = 0; }
  float curmax = 3.0e38f; int arg = base;
  const float* pb = points + (size_t)b * P_ * 3;
  const float* mb = mask   + (size_t)b * P_;
  for (int p = l; p < P_; p += 32) {
    float dx = pb[p*3+0]-kx, dy = pb[p*3+1]-ky, dz = pb[p*3+2]-kz;
    float d2 = dx*dx + dy*dy + dz*dz;
    if (mb[p] == 0.0f) d2 = 1.0e30f;                 // masked points pushed away
    if (d2 < curmax) {
      cd[arg] = d2; ci[arg] = p;
      curmax = cd[base]; arg = base;
#pragma unroll
      for (int j = 1; j < 16; ++j)
        if (cd[base + j] > curmax) { curmax = cd[base + j]; arg = base + j; }
    }
  }
  __syncthreads();
  for (int t = 0; t < 16; ++t) {
    float bd = 3.0e38f; int bs = base;
    for (int s = l; s < 512; s += 32)
      if (cd[s] < bd) { bd = cd[s]; bs = s; }
#pragma unroll
    for (int off = 16; off > 0; off >>= 1) {
      float od = __shfl_xor(bd, off, 32);
      int   os = __shfl_xor(bs, off, 32);
      if (od < bd) { bd = od; bs = os; }
    }
    if (l == 0) { idxOut[(size_t)bk*16 + t] = ci[bs]; cd[bs] = 3.0e38f; }
    __syncthreads();
  }
}

// ---------------------------------------------------------------------------
// Edge MLP + masked neighbor aggregation per key: e(64x64)=mp(64x16)@cf(16x64)
// via WMMA f16 (K padded 16->32 with zeros). Output E stored as bf16 row.
// ---------------------------------------------------------------------------
__global__ void edge_agg_kernel(const float* __restrict__ keys,
                                const float* __restrict__ points,
                                const float* __restrict__ feats,
                                const float* __restrict__ mask,
                                const int*   __restrict__ idxIn,
                                const float* __restrict__ w1, const float* __restrict__ b1,
                                const float* __restrict__ w2, const float* __restrict__ b2,
                                const float* __restrict__ w3, const float* __restrict__ b3,
                                __hip_bfloat16* __restrict__ E) {
  __shared__ float rel[N_][3];
  __shared__ float tm[N_];
  __shared__ int   ni[N_];
  __shared__ float h1[N_][H1_];
  __shared__ float h2[N_][H2_];
  __shared__ unsigned mA[CMID][16];   // A: 64 rows x 16 K-pairs (K=neighbors, padded)
  __shared__ unsigned mB[CIN][16];    // B^T: 64 cols x 16 K-pairs

  int bk = blockIdx.x; int b = bk >> 10;
  int tid = threadIdx.x; int lane = tid & 31; int wave = tid >> 5;

  if (tid < N_) ni[tid] = idxIn[(size_t)bk*N_ + tid];
  __syncthreads();
  if (tid < N_) {
    int p = ni[tid];
    const float* pp = points + ((size_t)b * P_ + p) * 3;
    rel[tid][0] = pp[0] - keys[(size_t)bk*3+0];
    rel[tid][1] = pp[1] - keys[(size_t)bk*3+1];
    rel[tid][2] = pp[2] - keys[(size_t)bk*3+2];
    tm[tid] = mask[(size_t)b * P_ + p];
  }
  __syncthreads();
  // layer 1: 3 -> 32, relu
  for (int t = tid; t < N_*H1_; t += 64) {
    int n = t >> 5, u = t & 31;
    float a = b1[u];
#pragma unroll
    for (int d = 0; d < 3; ++d) a += rel[n][d] * w1[d*H1_ + u];
    h1[n][u] = fmaxf(a, 0.0f);
  }
  __syncthreads();
  // layer 2: 32 -> 32, relu
  for (int t = tid; t < N_*H2_; t += 64) {
    int n = t >> 5, u = t & 31;
    float a = b2[u];
#pragma unroll
    for (int j = 0; j < H1_; ++j) a += h1[n][j] * w2[j*H2_ + u];
    h2[n][u] = fmaxf(a, 0.0f);
  }
  __syncthreads();
  // layer 3: 32 -> 64, relu, mask; pack as A-matrix f16 K-pairs
  for (int t = tid; t < CMID*8; t += 64) {
    int m = t >> 3, p = t & 7;
    float a0 = b3[m], a1 = b3[m];
#pragma unroll
    for (int j = 0; j < H2_; ++j) {
      float w = w3[j*CMID + m];
      a0 += h2[2*p+0][j] * w;
      a1 += h2[2*p+1][j] * w;
    }
    a0 = fmaxf(a0, 0.0f) * tm[2*p+0];
    a1 = fmaxf(a1, 0.0f) * tm[2*p+1];
    __half x0 = __float2half(a0), x1 = __float2half(a1);
    mA[m][p]   = (unsigned)__half_as_ushort(x0) | ((unsigned)__half_as_ushort(x1) << 16);
    mA[m][8+p] = 0u;                                 // K-pad 16..31
  }
  // gather features (masked) as B-matrix f16 K-pairs
  for (int t = tid; t < CIN*8; t += 64) {
    int c = t >> 3, p = t & 7;
    float f0 = feats[((size_t)b*P_ + ni[2*p+0])*CIN + c] * tm[2*p+0];
    float f1 = feats[((size_t)b*P_ + ni[2*p+1])*CIN + c] * tm[2*p+1];
    __half x0 = __float2half(f0), x1 = __float2half(f1);
    mB[c][p]   = (unsigned)__half_as_ushort(x0) | ((unsigned)__half_as_ushort(x1) << 16);
    mB[c][8+p] = 0u;
  }
  __syncthreads();

  int half = lane >> 4, lr = lane & 15;
  __hip_bfloat16* Erow = E + (size_t)bk * EDIM;
#pragma unroll
  for (int mt = 0; mt < 2; ++mt) {
    int Mtile = wave*2 + mt;
    v16h a = ldsFragH(&mA[Mtile*16 + lr][0], half);
#pragma unroll
    for (int nt = 0; nt < 4; ++nt) {
      v16h bb = ldsFragH(&mB[nt*16 + lr][0], half);
      v8f c = {};
      c = __builtin_amdgcn_wmma_f32_16x16x32_f16(false, a, false, bb,
                                                 (short)0, c, false, false);
#pragma unroll
      for (int r = 0; r < 8; ++r) {
        int mg = Mtile*16 + r + 8*half;              // C/D layout: M = r + 8*half
        Erow[mg*CIN + nt*16 + lr] = __float2bfloat16(c[r]);
      }
    }
  }
}

// ---------------------------------------------------------------------------
// GEMM1: H = relu(E(8192x4096) @ Wf1(4096x256) + bf1), bf16 WMMA, 128x64 tile.
// Double-buffered TDM pipeline: wave 0 issues tile k+1 while all waves compute
// tile k; s_wait_tensorcnt(2) = "oldest tile pair has landed".
// ---------------------------------------------------------------------------
__global__ void gemm1_kernel(const __hip_bfloat16* __restrict__ E,
                             const __hip_bfloat16* __restrict__ Wt1,
                             const float* __restrict__ bias,
                             __hip_bfloat16* __restrict__ Hout) {
  __shared__ unsigned As[2][128][16];   // ping-pong: 128 rows x 16 K-pairs
  __shared__ unsigned Bs[2][64][16];    // ping-pong: 64 cols x 16 K-pairs
  int Mb = blockIdx.x, Nb = blockIdx.y;
  int tid = threadIdx.x; int lane = tid & 31; int wave = tid >> 5;
  int half = lane >> 4, lr = lane & 15;
  const int NK = EDIM/32;
  v8f acc[4] = {};
#if HAVE_TDM
  unsigned ldsA[2] = { (unsigned)(size_t)&As[0][0][0], (unsigned)(size_t)&As[1][0][0] };
  unsigned ldsB[2] = { (unsigned)(size_t)&Bs[0][0][0], (unsigned)(size_t)&Bs[1][0][0] };
  const __hip_bfloat16* Abase = E   + (size_t)(Mb*128)*EDIM;
  const __hip_bfloat16* Bbase = Wt1 + (size_t)(Nb*64)*EDIM;
  if (wave == 0) {
    tdm_load_2d(ldsA[0], Abase, EDIM, ROWS, 32u, 128u, EDIM);
    tdm_load_2d(ldsB[0], Bbase, EDIM, FH_,  32u, 64u,  EDIM);
  }
  for (int kc = 0; kc < NK; ++kc) {
    int buf = kc & 1;
    if (wave == 0) {
      if (kc + 1 < NK) {
        int k1 = (kc + 1) * 32;
        // prefetch next tile pair into the other buffer (its last readers
        // passed the end-of-iteration barrier one iteration ago)
        tdm_load_2d(ldsA[buf ^ 1], Abase + k1, EDIM, ROWS, 32u, 128u, EDIM);
        tdm_load_2d(ldsB[buf ^ 1], Bbase + k1, EDIM, FH_,  32u, 64u,  EDIM);
        __builtin_amdgcn_s_wait_tensorcnt(2);   // oldest pair (current buf) done
      } else {
        __builtin_amdgcn_s_wait_tensorcnt(0);
      }
    }
    __syncthreads();                            // publish current buffer
    v16bf a = ldsFragBF(&As[buf][wave*16 + lr][0], half);
#pragma unroll
    for (int nt = 0; nt < 4; ++nt) {
      v16bf bb = ldsFragBF(&Bs[buf][nt*16 + lr][0], half);
      acc[nt] = __builtin_amdgcn_wmma_f32_16x16x32_bf16(false, a, false, bb,
                                                        (short)0, acc[nt], false, false);
    }
    __syncthreads();                            // all reads of buf complete
  }
#else
  for (int kc = 0; kc < NK; ++kc) {
    int k0 = kc * 32;
    for (int t = tid; t < 512; t += 256) {           // A: 512 x uint4
      int r = t >> 2, q = t & 3;
      const uint4* src = (const uint4*)(E + (size_t)(Mb*128 + r)*EDIM + k0 + q*8);
      *(uint4*)&As[0][r][q*4] = *src;
    }
    {                                                // B: 256 x uint4
      int n = tid >> 2, q = tid & 3;
      const uint4* src = (const uint4*)(Wt1 + (size_t)(Nb*64 + n)*EDIM + k0 + q*8);
      *(uint4*)&Bs[0][n][q*4] = *src;
    }
    __syncthreads();
    v16bf a = ldsFragBF(&As[0][wave*16 + lr][0], half);
#pragma unroll
    for (int nt = 0; nt < 4; ++nt) {
      v16bf bb = ldsFragBF(&Bs[0][nt*16 + lr][0], half);
      acc[nt] = __builtin_amdgcn_wmma_f32_16x16x32_bf16(false, a, false, bb,
                                                        (short)0, acc[nt], false, false);
    }
    __syncthreads();
    if (kc + 1 < NK && tid < 128)                    // gfx1250 global_prefetch_b8
      __builtin_prefetch(E + (size_t)(Mb*128 + tid)*EDIM + k0 + 32, 0, 1);
  }
#endif
#pragma unroll
  for (int nt = 0; nt < 4; ++nt) {
    int ng = Nb*64 + nt*16 + lr;
    float bv = bias[ng];
#pragma unroll
    for (int r = 0; r < 8; ++r) {
      int mg = Mb*128 + wave*16 + r + 8*half;
      Hout[(size_t)mg*FH_ + ng] = __float2bfloat16(fmaxf(acc[nt][r] + bv, 0.0f));
    }
  }
}

// ---------------------------------------------------------------------------
// GEMM2: out = H(8192x256) @ Wf2(256x128) + bf2  (fp32 output, no relu)
// ---------------------------------------------------------------------------
__global__ void gemm2_kernel(const __hip_bfloat16* __restrict__ Hin,
                             const __hip_bfloat16* __restrict__ Wt2,
                             const float* __restrict__ bias,
                             float* __restrict__ out) {
  __shared__ unsigned As[128][16];
  __shared__ unsigned Bs[64][16];
  int Mb = blockIdx.x, Nb = blockIdx.y;
  int tid = threadIdx.x; int lane = tid & 31; int wave = tid >> 5;
  int half = lane >> 4, lr = lane & 15;
  v8f acc[4] = {};
  for (int kc = 0; kc < FH_/32; ++kc) {
    int k0 = kc * 32;
    for (int t = tid; t < 512; t += 256) {
      int r = t >> 2, q = t & 3;
      const uint4* src = (const uint4*)(Hin + (size_t)(Mb*128 + r)*FH_ + k0 + q*8);
      *(uint4*)&As[r][q*4] = *src;
    }
    {
      int n = tid >> 2, q = tid & 3;
      const uint4* src = (const uint4*)(Wt2 + (size_t)(Nb*64 + n)*FH_ + k0 + q*8);
      *(uint4*)&Bs[n][q*4] = *src;
    }
    __syncthreads();
    v16bf a = ldsFragBF(&As[wave*16 + lr][0], half);
#pragma unroll
    for (int nt = 0; nt < 4; ++nt) {
      v16bf bb = ldsFragBF(&Bs[nt*16 + lr][0], half);
      acc[nt] = __builtin_amdgcn_wmma_f32_16x16x32_bf16(false, a, false, bb,
                                                        (short)0, acc[nt], false, false);
    }
    __syncthreads();
  }
#pragma unroll
  for (int nt = 0; nt < 4; ++nt) {
    int ng = Nb*64 + nt*16 + lr;
    float bv = bias[ng];
#pragma unroll
    for (int r = 0; r < 8; ++r) {
      int mg = Mb*128 + wave*16 + r + 8*half;
      out[(size_t)mg*COUT + ng] = acc[nt][r] + bv;
    }
  }
}

// ---------------------------------------------------------------------------
extern "C" void kernel_launch(void* const* d_in, const int* in_sizes, int n_in,
                              void* d_out, int out_size, void* d_ws, size_t ws_size,
                              hipStream_t stream) {
  const float* keys   = (const float*)d_in[0];
  const float* points = (const float*)d_in[1];
  const float* feats  = (const float*)d_in[2];
  const float* pmask  = (const float*)d_in[3];
  const float* w1 = (const float*)d_in[4];  const float* b1 = (const float*)d_in[5];
  const float* w2 = (const float*)d_in[6];  const float* b2 = (const float*)d_in[7];
  const float* w3 = (const float*)d_in[8];  const float* b3 = (const float*)d_in[9];
  const float* wf1 = (const float*)d_in[10]; const float* bf1 = (const float*)d_in[11];
  const float* wf2 = (const float*)d_in[12]; const float* bf2 = (const float*)d_in[13];
  (void)in_sizes; (void)n_in; (void)out_size; (void)ws_size;

  char* ws = (char*)d_ws;
  size_t off = 0;
  int* idx = (int*)(ws + off);                 off += (size_t)ROWS * N_ * sizeof(int);
  off = (off + 255) & ~(size_t)255;
  __hip_bfloat16* E   = (__hip_bfloat16*)(ws + off); off += (size_t)ROWS * EDIM * 2;
  __hip_bfloat16* H   = (__hip_bfloat16*)(ws + off); off += (size_t)ROWS * FH_ * 2;
  __hip_bfloat16* Wt1 = (__hip_bfloat16*)(ws + off); off += (size_t)EDIM * FH_ * 2;
  __hip_bfloat16* Wt2 = (__hip_bfloat16*)(ws + off); off += (size_t)FH_ * COUT * 2;

  int prepTot = EDIM*FH_ + FH_*COUT;
  prep_weights<<<(prepTot + 255)/256, 256, 0, stream>>>(wf1, wf2, Wt1, Wt2);
  topk_kernel<<<ROWS, 32, 0, stream>>>(keys, points, pmask, idx);
  edge_agg_kernel<<<ROWS, 64, 0, stream>>>(keys, points, feats, pmask, idx,
                                           w1, b1, w2, b2, w3, b3, E);
  gemm1_kernel<<<dim3(ROWS/128, FH_/64), 256, 0, stream>>>(E, Wt1, bf1, H);
  gemm2_kernel<<<dim3(ROWS/128, COUT/64), 256, 0, stream>>>(H, Wt2, bf2, (float*)d_out);
}
